// CrossAttention_42382737277217
// MI455X (gfx1250) — compile-verified, matrix operand order checked by reading, not run on hardware
//
#include <hip/hip_runtime.h>
#include <hip/hip_bf16.h>
#include <math.h>

typedef _Float16 f16;
typedef __attribute__((ext_vector_type(8)))  _Float16 v8h;
typedef __attribute__((ext_vector_type(16))) _Float16 v16h;
typedef __attribute__((ext_vector_type(8)))  float    v8f;

#define HNUM 16
#define DH   64
#define CDIM 1024
#define NSEQ 2048
#define BDIM 2

#define ATTN_SCALE (0.125f / 1.5f)   // Dh^-0.5 / ATTN_TEMP
#define CLS_SCALE  (0.125f / 2.0f)   // Dh^-0.5 / CLS_TEMP
#define INV_H      (1.0f / 16.0f)

// ---------------------------------------------------------------------------
// gfx1250 async global->LDS staging via inline asm (GLOBAL_LOAD_ASYNC_TO_LDS_
// B128, tracked by ASYNCcnt).  VDST = 32-bit LDS byte address (low 32 bits of
// the flat pointer per the LDS aperture rules), VADDR = 64-bit global address.
// The instruction offset is added to BOTH addresses (ISA §8 async pseudocode),
// so offset:0/offset:16 copies 32 bytes per lane from one address pair.
// ---------------------------------------------------------------------------
static __device__ __forceinline__ void async_copy32B(f16* l, const f16* g) {
  unsigned lds = (unsigned)(unsigned long long)l;   // LDS byte offset
  asm volatile(
      "global_load_async_to_lds_b128 %0, %1, off\n\t"
      "global_load_async_to_lds_b128 %0, %1, off offset:16"
      :: "v"(lds), "v"(g) : "memory");
}
static __device__ __forceinline__ void async_wait0() {
  asm volatile("s_wait_asynccnt 0" ::: "memory");
}

// ---------------------------------------------------------------------------
// Fragment loader for 16x16x32 f16 WMMA, wave32 (CDNA5 ISA layouts).
// A operand of a row-major [16 x >=32] tile; equally the B operand when the
// tile is stored N-major (tile[n][k]).  Two contiguous 16-byte chunks ->
// 2 x ds_load_b128 / global_load_b128.
// ---------------------------------------------------------------------------
static __device__ __forceinline__ v16h load_frag(const f16* p0, int ld, int lane) {
  const f16* p = p0 + (size_t)(lane & 15) * ld + ((lane >> 4) & 1) * 8;
  v8h lo = *(const v8h*)p;
  v8h hi = *(const v8h*)(p + 16);
  return __builtin_shufflevector(lo, hi, 0, 1, 2, 3, 4, 5, 6, 7,
                                 8, 9, 10, 11, 12, 13, 14, 15);
}

static __device__ __forceinline__ v8f wmma16(v16h a, v16h b, v8f c) {
  return __builtin_amdgcn_wmma_f32_16x16x32_f16(false, a, false, b, (short)0, c,
                                                false, false);
}

static __device__ __forceinline__ v8h cvt8(float4 a, float4 b) {
  v8h h;
  h[0] = (f16)a.x; h[1] = (f16)a.y; h[2] = (f16)a.z; h[3] = (f16)a.w;
  h[4] = (f16)b.x; h[5] = (f16)b.y; h[6] = (f16)b.z; h[7] = (f16)b.w;
  return h;
}

// ---------------------------------------------------------------------------
// Kernel 1: Q/K/V projections.  out[n,o] = x[n,:]·W[o,:] + bias[o]
// 64x64 workgroup tile, 8 waves, K-step 64.  Both LDS tiles are row-major in
// the "lane dimension" so every fragment load is 2 x ds_load_b128.
// (Synchronous staging: must convert f32->f16 in flight.)
// grid: (rows/64, C/64, 3)
// ---------------------------------------------------------------------------
__global__ __launch_bounds__(256) void qkv_gemm(
    const float* __restrict__ xq, const float* __restrict__ xkv,
    const float* __restrict__ Wq, const float* __restrict__ bq,
    const float* __restrict__ Wk, const float* __restrict__ bk,
    const float* __restrict__ Wv, const float* __restrict__ bv,
    f16* __restrict__ Qh, f16* __restrict__ Kh, f16* __restrict__ Vh)
{
  __shared__ f16 At[64 * 64];   // A rows x K        (8 KB)
  __shared__ f16 Wt[64 * 64];   // out-cols x K      (8 KB)

  const int z = blockIdx.z;
  const float* X    = (z == 0) ? xq : xkv;
  const float* W    = (z == 0) ? Wq : ((z == 1) ? Wk : Wv);
  const float* bias = (z == 0) ? bq : ((z == 1) ? bk : bv);
  f16* Out          = (z == 0) ? Qh : ((z == 1) ? Kh : Vh);

  const int row0 = blockIdx.x * 64;
  const int o0   = blockIdx.y * 64;
  const int tid  = threadIdx.x;
  const int lane = tid & 31;
  const int w    = tid >> 5;
  const int fr   = w >> 1;          // 0..3 : 16-row stripe
  const int fc0  = (w & 1) * 2;     // 0 or 2 : first of two 16-col stripes
  const int lr   = tid >> 2;        // 0..63 : staging row
  const int lc   = (tid & 3) * 16;  // staging col chunk (16 elems)
  v8f acc0 = {}, acc1 = {};

  for (int kk = 0; kk < CDIM; kk += 64) {
    {
      const float* sa = X + (size_t)(row0 + lr) * CDIM + kk + lc;
      float4 a0 = *(const float4*)(sa);
      float4 a1 = *(const float4*)(sa + 4);
      float4 a2 = *(const float4*)(sa + 8);
      float4 a3 = *(const float4*)(sa + 12);
      const float* sw = W + (size_t)(o0 + lr) * CDIM + kk + lc;
      float4 w0 = *(const float4*)(sw);
      float4 w1 = *(const float4*)(sw + 4);
      float4 w2 = *(const float4*)(sw + 8);
      float4 w3 = *(const float4*)(sw + 12);
      *(v8h*)(At + lr * 64 + lc)     = cvt8(a0, a1);
      *(v8h*)(At + lr * 64 + lc + 8) = cvt8(a2, a3);
      *(v8h*)(Wt + lr * 64 + lc)     = cvt8(w0, w1);
      *(v8h*)(Wt + lr * 64 + lc + 8) = cvt8(w2, w3);
    }
    __syncthreads();
    v16h a0 = load_frag(At + fr * 16 * 64,      64, lane);
    v16h a1 = load_frag(At + fr * 16 * 64 + 32, 64, lane);
    v16h b00 = load_frag(Wt + (fc0 * 16) * 64,            64, lane);
    v16h b01 = load_frag(Wt + (fc0 * 16) * 64 + 32,       64, lane);
    v16h b10 = load_frag(Wt + ((fc0 + 1) * 16) * 64,      64, lane);
    v16h b11 = load_frag(Wt + ((fc0 + 1) * 16) * 64 + 32, 64, lane);
    acc0 = wmma16(a0, b00, acc0);
    acc0 = wmma16(a1, b01, acc0);
    acc1 = wmma16(a0, b10, acc1);
    acc1 = wmma16(a1, b11, acc1);
    __syncthreads();
  }

  const int n  = lane & 15;
  const int mb = (lane >> 4) * 8;
#pragma unroll
  for (int g = 0; g < 8; ++g) {
    int R  = row0 + fr * 16 + mb + g;
    int bI = R >> 11;                // / NSEQ
    int nn = R & (NSEQ - 1);
    int oA = o0 + fc0 * 16 + n;
    int oB = oA + 16;
    float vA = acc0[g] + bias[oA];
    float vB = acc1[g] + bias[oB];
    Out[((size_t)((bI * HNUM + (oA >> 6)) * NSEQ + nn)) * DH + (oA & 63)] = (f16)vA;
    Out[((size_t)((bI * HNUM + (oB >> 6)) * NSEQ + nn)) * DH + (oB & 63)] = (f16)vB;
  }
}

// ---------------------------------------------------------------------------
// Kernel 2: affinity mask.  aff[k] = mean_h sigmoid(cls_h · k_h * CLS_SCALE),
// then min/max normalize over Nk.  One block per batch.
// ---------------------------------------------------------------------------
__global__ __launch_bounds__(256) void aff_kernel(
    const float* __restrict__ cls, const f16* __restrict__ Kh,
    float* __restrict__ affW, float* __restrict__ affOut)
{
  __shared__ float rmin[256], rmax[256];
  const int b = blockIdx.x;
  const int tid = threadIdx.x;

  float vals[8];
  float lmin = __builtin_inff(), lmax = -__builtin_inff();
#pragma unroll
  for (int i = 0; i < 8; ++i) {
    int k = tid + i * 256;
    float s = 0.f;
    for (int h = 0; h < HNUM; ++h) {
      const float* c = cls + (size_t)b * CDIM + h * DH;
      const f16* kp = Kh + ((size_t)(b * HNUM + h) * NSEQ + k) * DH;
      float dot = 0.f;
#pragma unroll
      for (int d8 = 0; d8 < 8; ++d8) {
        v8h kv = *(const v8h*)(kp + d8 * 8);
        const float* cc = c + d8 * 8;
#pragma unroll
        for (int j = 0; j < 8; ++j) dot += cc[j] * (float)kv[j];
      }
      s += 1.f / (1.f + __expf(-dot * CLS_SCALE));
    }
    s *= INV_H;
    vals[i] = s;
    lmin = fminf(lmin, s);
    lmax = fmaxf(lmax, s);
  }
  rmin[tid] = lmin; rmax[tid] = lmax;
  __syncthreads();
  for (int st = 128; st > 0; st >>= 1) {
    if (tid < st) {
      rmin[tid] = fminf(rmin[tid], rmin[tid + st]);
      rmax[tid] = fmaxf(rmax[tid], rmax[tid + st]);
    }
    __syncthreads();
  }
  const float m = rmin[0], M = rmax[0];
  const float inv = 1.f / (M - m + 1e-6f);
#pragma unroll
  for (int i = 0; i < 8; ++i) {
    int k = tid + i * 256;
    float nv = (vals[i] - m) * inv;
    affW[b * NSEQ + k]   = nv;
    affOut[b * NSEQ + k] = nv;
  }
}

// ---------------------------------------------------------------------------
// Kernel 3: flash cross-attention.  Block = (b, 64 q-rows), 4 waves.
// Heads looped sequentially inside the block so the head-summed logits output
// is a private read-modify-write (no atomics).  K tile staged row-major via
// async global->LDS DMA (S B-operand = A-pattern); V tile staged transposed
// (P·V B-operand = A-pattern).  All fragment reads are 2 x ds_load_b128.
// ---------------------------------------------------------------------------
__global__ __launch_bounds__(128) void attn_kernel(
    const f16* __restrict__ Qh, const f16* __restrict__ Kh,
    const f16* __restrict__ Vh, const float* __restrict__ affW,
    float* __restrict__ logitsOut, f16* __restrict__ Xa)
{
  __shared__ float affL[NSEQ];       // 8 KB
  __shared__ f16 Kt[32 * 64];        // 4 KB, [k][d]
  __shared__ f16 VtT[64 * 32];       // 4 KB, [d][k]  (transposed)
  __shared__ f16 Pt[4][16 * 32];     // 4 KB (per-wave P staging)

  const int blk  = blockIdx.x;
  const int b    = blk >> 5;
  const int q0   = (blk & 31) * 64;
  const int tid  = threadIdx.x;
  const int lane = tid & 31;
  const int w    = tid >> 5;
  const int qw   = q0 + w * 16;
  const int n    = lane & 15;
  const int mb   = (lane >> 4) * 8;
  const int lk   = tid >> 2;         // 0..31 : k row for staging
  const int ld0  = (tid & 3) * 16;   // d chunk (16 elems)

  for (int e = tid; e < NSEQ; e += 128) affL[e] = affW[b * NSEQ + e];
  __syncthreads();

  for (int h = 0; h < HNUM; ++h) {
    const f16* qbase = Qh + ((size_t)((b * HNUM + h) * NSEQ + qw)) * DH;
    const f16* kbase = Kh + ((size_t)(b * HNUM + h) * NSEQ) * DH;
    const f16* vbase = Vh + ((size_t)(b * HNUM + h) * NSEQ) * DH;

    v16h qa0 = load_frag(qbase,      DH, lane);   // d = 0..31
    v16h qa1 = load_frag(qbase + 32, DH, lane);   // d = 32..63

    float m8[8], l8[8];
    v8f o0 = {}, o1 = {}, o2 = {}, o3 = {};
#pragma unroll
    for (int g = 0; g < 8; ++g) { m8[g] = -__builtin_inff(); l8[g] = 0.f; }

    for (int kt = 0; kt < NSEQ; kt += 32) {
      // ---- stage K (row-major, async DMA to LDS) and V (transposed, manual)
      {
        const f16* ks = kbase + (size_t)(kt + lk) * DH + ld0;
        f16* kd = Kt + lk * 64 + ld0;
        async_copy32B(kd, ks);      // 2 x GLOBAL_LOAD_ASYNC_TO_LDS_B128

        const f16* vs = vbase + (size_t)(kt + lk) * DH + ld0;
        v8h v0 = *(const v8h*)(vs);
        v8h v1 = *(const v8h*)(vs + 8);
#pragma unroll
        for (int j = 0; j < 8; ++j) {
          VtT[(ld0 + j) * 32 + lk]     = v0[j];
          VtT[(ld0 + 8 + j) * 32 + lk] = v1[j];
        }
        if (kt + 32 < NSEQ) {
          __builtin_prefetch(kbase + (size_t)(kt + 32 + lk) * DH + ld0, 0, 1);
          __builtin_prefetch(vbase + (size_t)(kt + 32 + lk) * DH + ld0, 0, 1);
        }
        async_wait0();              // ASYNCcnt==0 before the barrier
      }
      __syncthreads();

      // ---- S = Q·K^T (B-operand = A-pattern on row-major Kt)
      v8f s0 = {}, s1 = {};
      s0 = wmma16(qa0, load_frag(Kt,                DH, lane), s0);
      s0 = wmma16(qa1, load_frag(Kt + 32,           DH, lane), s0);
      s1 = wmma16(qa0, load_frag(Kt + 16 * DH,      DH, lane), s1);
      s1 = wmma16(qa1, load_frag(Kt + 16 * DH + 32, DH, lane), s1);
#pragma unroll
      for (int g = 0; g < 8; ++g) { s0[g] *= ATTN_SCALE; s1[g] *= ATTN_SCALE; }

      // ---- accumulate head-summed logits output (/H); h==0 initializes
      {
        float* lrow = logitsOut + ((size_t)(b * NSEQ + qw + mb)) * NSEQ + kt + n;
#pragma unroll
        for (int g = 0; g < 8; ++g) {
          float* p = lrow + (size_t)g * NSEQ;
          float v0 = s0[g] * INV_H, v1 = s1[g] * INV_H;
          if (h == 0) { p[0] = v0;  p[16] = v1; }
          else        { p[0] += v0; p[16] += v1; }
        }
      }

      // ---- online softmax (row stats across 16 lanes of each half-wave)
      const float a0 = affL[kt + n];
      const float a1 = affL[kt + 16 + n];
      float corr[8];
#pragma unroll
      for (int g = 0; g < 8; ++g) {
        float t = fmaxf(s0[g], s1[g]);
        t = fmaxf(t, __shfl_xor(t, 1, 32));
        t = fmaxf(t, __shfl_xor(t, 2, 32));
        t = fmaxf(t, __shfl_xor(t, 4, 32));
        t = fmaxf(t, __shfl_xor(t, 8, 32));
        float nm = fmaxf(m8[g], t);
        corr[g] = __expf(m8[g] - nm);
        float p0 = __expf(s0[g] - nm);
        float p1 = __expf(s1[g] - nm);
        float rs = p0 + p1;
        rs += __shfl_xor(rs, 1, 32);
        rs += __shfl_xor(rs, 2, 32);
        rs += __shfl_xor(rs, 4, 32);
        rs += __shfl_xor(rs, 8, 32);
        l8[g] = l8[g] * corr[g] + rs;
        m8[g] = nm;
        // mask applied post-softmax => weight P, not the denominator
        Pt[w][(mb + g) * 32 + n]      = (f16)(p0 * a0);
        Pt[w][(mb + g) * 32 + 16 + n] = (f16)(p1 * a1);
      }
#pragma unroll
      for (int g = 0; g < 8; ++g) {
        o0[g] *= corr[g]; o1[g] *= corr[g]; o2[g] *= corr[g]; o3[g] *= corr[g];
      }
      __syncthreads();   // P staging visible (uniform across waves)

      // ---- O += P·V (B-operand = A-pattern on transposed VtT)
      v16h pa = load_frag(&Pt[w][0], 32, lane);
      o0 = wmma16(pa, load_frag(VtT,           32, lane), o0);
      o1 = wmma16(pa, load_frag(VtT + 16 * 32, 32, lane), o1);
      o2 = wmma16(pa, load_frag(VtT + 32 * 32, 32, lane), o2);
      o3 = wmma16(pa, load_frag(VtT + 48 * 32, 32, lane), o3);
      __syncthreads();   // before next K/V tile overwrite
    }

    // ---- finalize: O /= l, write f16 attn output laid out [B, Nq, C]
#pragma unroll
    for (int g = 0; g < 8; ++g) {
      float inv = 1.f / l8[g];
      int R = qw + mb + g;
      size_t base = ((size_t)(b * NSEQ + R)) * CDIM + h * DH + n;
      Xa[base]      = (f16)(o0[g] * inv);
      Xa[base + 16] = (f16)(o1[g] * inv);
      Xa[base + 32] = (f16)(o2[g] * inv);
      Xa[base + 48] = (f16)(o3[g] * inv);
    }
  }
}

// ---------------------------------------------------------------------------
// Kernel 4: output projection.  out = Xa·Wp^T + bp  (f32 out)
// A tile is raw f16 -> staged with async global->LDS DMA.
// ---------------------------------------------------------------------------
__global__ __launch_bounds__(256) void out_gemm(
    const f16* __restrict__ Xa, const float* __restrict__ Wp,
    const float* __restrict__ bp, float* __restrict__ Out)
{
  __shared__ f16 At[64 * 64];
  __shared__ f16 Wt[64 * 64];

  const int row0 = blockIdx.x * 64;
  const int o0   = blockIdx.y * 64;
  const int tid  = threadIdx.x;
  const int lane = tid & 31;
  const int w    = tid >> 5;
  const int fr   = w >> 1;
  const int fc0  = (w & 1) * 2;
  const int lr   = tid >> 2;
  const int lc   = (tid & 3) * 16;
  v8f acc0 = {}, acc1 = {};

  for (int kk = 0; kk < CDIM; kk += 64) {
    {
      const f16* sa = Xa + (size_t)(row0 + lr) * CDIM + kk + lc;
      f16* da = At + lr * 64 + lc;
      async_copy32B(da, sa);        // 2 x GLOBAL_LOAD_ASYNC_TO_LDS_B128

      const float* sw = Wp + (size_t)(o0 + lr) * CDIM + kk + lc;
      float4 w0 = *(const float4*)(sw);
      float4 w1 = *(const float4*)(sw + 4);
      float4 w2 = *(const float4*)(sw + 8);
      float4 w3 = *(const float4*)(sw + 12);
      *(v8h*)(Wt + lr * 64 + lc)     = cvt8(w0, w1);
      *(v8h*)(Wt + lr * 64 + lc + 8) = cvt8(w2, w3);
      async_wait0();
    }
    __syncthreads();
    v16h a0 = load_frag(At + fr * 16 * 64,      64, lane);
    v16h a1 = load_frag(At + fr * 16 * 64 + 32, 64, lane);
    acc0 = wmma16(a0, load_frag(Wt + (fc0 * 16) * 64,            64, lane), acc0);
    acc0 = wmma16(a1, load_frag(Wt + (fc0 * 16) * 64 + 32,       64, lane), acc0);
    acc1 = wmma16(a0, load_frag(Wt + ((fc0 + 1) * 16) * 64,      64, lane), acc1);
    acc1 = wmma16(a1, load_frag(Wt + ((fc0 + 1) * 16) * 64 + 32, 64, lane), acc1);
    __syncthreads();
  }

  const int n  = lane & 15;
  const int mb = (lane >> 4) * 8;
#pragma unroll
  for (int g = 0; g < 8; ++g) {
    int R  = row0 + fr * 16 + mb + g;
    int oA = o0 + fc0 * 16 + n;
    int oB = oA + 16;
    Out[(size_t)R * CDIM + oA] = acc0[g] + bp[oA];
    Out[(size_t)R * CDIM + oB] = acc1[g] + bp[oB];
  }
}

// ---------------------------------------------------------------------------
extern "C" void kernel_launch(void* const* d_in, const int* in_sizes, int n_in,
                              void* d_out, int out_size, void* d_ws, size_t ws_size,
                              hipStream_t stream) {
  (void)in_sizes; (void)n_in; (void)out_size; (void)ws_size;
  const float* x_q  = (const float*)d_in[0];
  const float* x_kv = (const float*)d_in[1];
  const float* cls  = (const float*)d_in[2];
  const float* Wq   = (const float*)d_in[3];
  const float* bq   = (const float*)d_in[4];
  const float* Wk   = (const float*)d_in[5];
  const float* bk   = (const float*)d_in[6];
  const float* Wv   = (const float*)d_in[7];
  const float* bv   = (const float*)d_in[8];
  const float* Wp   = (const float*)d_in[9];
  const float* bp   = (const float*)d_in[10];
  float* out = (float*)d_out;

  const size_t PER = (size_t)BDIM * HNUM * NSEQ * DH;  // 4,194,304 elems
  f16* Qh = (f16*)d_ws;
  f16* Kh = Qh + PER;
  f16* Vh = Kh + PER;
  f16* Xa = Vh + PER;                                  // [B, Nq, C] f16
  float* affW = (float*)(Xa + PER);                    // [B, Nk] f32

  float* xOut      = out;                                    // [B,Nq,C]
  float* logitsOut = out + (size_t)BDIM * NSEQ * CDIM;       // [B,Nq,Nk]
  float* affOut    = logitsOut + (size_t)BDIM * NSEQ * NSEQ; // [B,Nk]

  qkv_gemm<<<dim3(64, 16, 3), 256, 0, stream>>>(x_q, x_kv, Wq, bq, Wk, bk,
                                                Wv, bv, Qh, Kh, Vh);
  aff_kernel<<<dim3(BDIM), 256, 0, stream>>>(cls, Kh, affW, affOut);
  attn_kernel<<<dim3(64), 128, 0, stream>>>(Qh, Kh, Vh, affW, logitsOut, Xa);
  out_gemm<<<dim3(64, 16), 256, 0, stream>>>(Xa, Wp, bp, xOut);
}
